// ImprovedSelfAttention_50044958933059
// MI455X (gfx1250) — matrix operand order repaired and checked by reading, not compiled, not run on hardware
//
#include <hip/hip_runtime.h>

typedef __attribute__((ext_vector_type(16))) __bf16   v16bf;
typedef __attribute__((ext_vector_type(8)))  float    v8f;
typedef __attribute__((ext_vector_type(4)))  unsigned u32x4;
typedef __attribute__((ext_vector_type(2)))  unsigned u32x2;
typedef __attribute__((ext_vector_type(4)))  float    f32x4;
typedef __attribute__((ext_vector_type(2)))  float    f32x2;

#define B_ 8
#define T_ 2048
#define C_ 512
#define U_ 256

// ---- packed f32 -> 2x bf16 via documented V_FMA_MIXLO/HI_BF16 (CDNA5 §15.10)
__device__ __forceinline__ unsigned pk2(float lo, float hi) {
    unsigned d;
    asm("v_fma_mixlo_bf16 %0, %1, 1.0, 0\n\t"
        "v_fma_mixhi_bf16 %0, %2, 1.0, 0"
        : "=&v"(d)
        : "v"(lo), "v"(hi));
    return d;
}
__device__ __forceinline__ float fast_rcp(float x) {
#if __has_builtin(__builtin_amdgcn_rcpf)
    return __builtin_amdgcn_rcpf(x);
#else
    return 1.0f / x;
#endif
}
__device__ __forceinline__ float fast_exp2(float x) {
#if __has_builtin(__builtin_amdgcn_exp2f)
    return __builtin_amdgcn_exp2f(x);   // v_exp_f32 is natively 2^x
#else
    return exp2f(x);
#endif
}

// ---- fragment load: two 16B LDS chunks -> v16bf ----------------------------
union V16U { v16bf v; u32x4 q[2]; };
__device__ __forceinline__ v16bf load_frag16(const void* p0, const void* p1) {
    V16U u;
    u.q[0] = *(const u32x4*)p0;
    u.q[1] = *(const u32x4*)p1;
    return u.v;
}

// ---- CDNA5 async copy: global -> LDS, 16 bytes per lane --------------------
__device__ __forceinline__ void async_copy_b128(unsigned lds_off, const void* gptr) {
    asm volatile("global_load_async_to_lds_b128 %0, %1, off"
                 :: "v"(lds_off), "v"(gptr) : "memory");
}
__device__ __forceinline__ void wait_asynccnt0() {
    asm volatile("s_wait_asynccnt 0x0" ::: "memory");
}
__device__ __forceinline__ unsigned lds_offset(const void* p) {
    return (unsigned)(unsigned long long)p;   // LDS generic addr: low 32b = offset
}

// ---------------------------------------------------------------------------
// Kernel 1: QKV projection  [BT,C] x [C,U] + bias -> bf16 [BT,U]
// ---------------------------------------------------------------------------
__global__ __launch_bounds__(256)
void qkv_proj_kernel(const float* __restrict__ x, const float* __restrict__ W,
                     const float* __restrict__ bias, unsigned short* __restrict__ out)
{
    __shared__ unsigned short ldsX[64][40];   // 80B rows (16B aligned)
    __shared__ unsigned short ldsWT[32][40];  // W transposed: [col][k]
    const int tid = threadIdx.x;
    const int lane = tid & 31, wave = tid >> 5;
    const int half = lane >> 4, lr = lane & 15;
    const int rowBase = blockIdx.x * 64;
    const int colBase = blockIdx.y * 32;
    const int wRow = (wave & 3) * 16;
    const int wCol = (wave >> 2) * 16;

    v8f acc = {0.f, 0.f, 0.f, 0.f, 0.f, 0.f, 0.f, 0.f};

    for (int k0 = 0; k0 < C_; k0 += 32) {
        // stage x tile (64x32) as bf16: float4 loads + packed converts
        #pragma unroll
        for (int ii = 0; ii < 2; ++ii) {
            int ch = tid + ii * 256;
            int r = ch >> 3, c4 = (ch & 7) * 4;
            f32x4 xv = *(const f32x4*)&x[(size_t)(rowBase + r) * C_ + k0 + c4];
            u32x2 o;
            o[0] = pk2(xv[0], xv[1]);
            o[1] = pk2(xv[2], xv[3]);
            *(u32x2*)&ldsX[r][c4] = o;
        }
        // stage W tile transposed, paired converts
        #pragma unroll
        for (int ii = 0; ii < 2; ++ii) {
            int idx = tid + ii * 256;
            int r = idx >> 4, c2 = (idx & 15) * 2;
            f32x2 wv = *(const f32x2*)&W[(size_t)(k0 + r) * U_ + colBase + c2];
            unsigned u = pk2(wv[0], wv[1]);
            ldsWT[c2][r]     = (unsigned short)u;
            ldsWT[c2 + 1][r] = (unsigned short)(u >> 16);
        }
        __syncthreads();

        v16bf a = load_frag16(&ldsX[wRow + lr][half * 8], &ldsX[wRow + lr][16 + half * 8]);
        v16bf b = load_frag16(&ldsWT[wCol + lr][half * 8], &ldsWT[wCol + lr][16 + half * 8]);
        acc = __builtin_amdgcn_wmma_f32_16x16x32_bf16(false, a, false, b,
                                                      (short)0, acc, false, false);
        __syncthreads();
    }

    const int col = colBase + wCol + lr;
    const float bval = bias[col];
    #pragma unroll
    for (int r = 0; r < 8; r += 2) {
        int row = rowBase + wRow + half * 8 + r;
        unsigned u = pk2(acc[r] + bval, acc[r + 1] + bval);
        out[(size_t)row * U_ + col]       = (unsigned short)u;
        out[(size_t)(row + 1) * U_ + col] = (unsigned short)(u >> 16);
    }
}

// ---------------------------------------------------------------------------
// Kernel 2: flash attention, double-buffered K/V staging.
// grid = (T/128, B); 8 waves, 16 Q rows per wave
// ---------------------------------------------------------------------------
__global__ __launch_bounds__(256)
void attn_kernel(const unsigned short* __restrict__ q,
                 const unsigned short* __restrict__ k,
                 const unsigned short* __restrict__ v,
                 unsigned short* __restrict__ aout)
{
    __shared__ unsigned short ldsK[2][32][U_];    // [buf][key][u]  2x16 KB (async)
    __shared__ unsigned short ldsVT[2][U_][32];   // [buf][u][key]  2x16 KB
    __shared__ unsigned short ldsP[8][16][32];    // per-wave P staging, 8 KB

    const int tid = threadIdx.x, lane = tid & 31, wave = tid >> 5;
    const int half = lane >> 4, lr = lane & 15;
    const int b = blockIdx.y;
    const int qRow0 = blockIdx.x * 128 + wave * 16;

    const unsigned short* qb = q + (size_t)b * T_ * U_;
    const unsigned short* kb = k + (size_t)b * T_ * U_;
    const unsigned short* vb = v + (size_t)b * T_ * U_;

    // per-thread copy coordinates (same mapping every chunk)
    int crow[4], ccol[4];
    #pragma unroll
    for (int ii = 0; ii < 4; ++ii) {
        int ch = tid + ii * 256;
        crow[ii] = ch >> 5;
        ccol[ii] = (ch & 31) * 8;
    }

    // preload Q: 8 A-fragments over U=256, b128 global loads
    v16bf qf[8];
    #pragma unroll
    for (int j = 0; j < 8; ++j) {
        const unsigned short* base = qb + (size_t)(qRow0 + lr) * U_ + j * 32;
        qf[j] = load_frag16(base + half * 8, base + 16 + half * 8);
    }

    v8f acc[16];
    #pragma unroll
    for (int t = 0; t < 16; ++t)
        acc[t] = v8f{0.f, 0.f, 0.f, 0.f, 0.f, 0.f, 0.f, 0.f};
    float mrow[8], lrow[8];
    #pragma unroll
    for (int r = 0; r < 8; ++r) { mrow[r] = -1e30f; lrow[r] = 0.f; }

    const float kScale = 0.0625f * 1.44269504089f;  // (1/sqrt(U)) * log2(e)

    // ---- prologue: start chunk 0 (async K into buf 0, V chunk into regs) ---
    #pragma unroll
    for (int ii = 0; ii < 4; ++ii)
        async_copy_b128(lds_offset(&ldsK[0][crow[ii]][ccol[ii]]),
                        kb + (size_t)crow[ii] * U_ + ccol[ii]);
    u32x4 vhold[4];
    #pragma unroll
    for (int ii = 0; ii < 4; ++ii)
        vhold[ii] = *(const u32x4*)&vb[(size_t)crow[ii] * U_ + ccol[ii]];

    for (int s0 = 0; s0 < T_; s0 += 32) {
        const int buf = (s0 >> 5) & 1;

        // commit held V registers (this chunk) into ldsVT[buf], transposed
        #pragma unroll
        for (int ii = 0; ii < 4; ++ii) {
            union { u32x4 q; unsigned short s[8]; } d;
            d.q = vhold[ii];
            #pragma unroll
            for (int e = 0; e < 8; ++e)
                ldsVT[buf][ccol[ii] + e][crow[ii]] = d.s[e];
        }
        wait_asynccnt0();    // this chunk's async K copies done
        __syncthreads();     // staging visible; all waves past previous compute

        // kick off next chunk into the other buffer (overlaps compute below)
        if (s0 + 32 < T_) {
            #pragma unroll
            for (int ii = 0; ii < 4; ++ii)
                async_copy_b128(lds_offset(&ldsK[buf ^ 1][crow[ii]][ccol[ii]]),
                                kb + (size_t)(s0 + 32 + crow[ii]) * U_ + ccol[ii]);
            #pragma unroll
            for (int ii = 0; ii < 4; ++ii)
                vhold[ii] = *(const u32x4*)&vb[(size_t)(s0 + 32 + crow[ii]) * U_ + ccol[ii]];
        }

        // scores for keys [s0, s0+32): two 16-key D tiles
        v8f S0v = {0.f,0.f,0.f,0.f,0.f,0.f,0.f,0.f};
        v8f S1v = {0.f,0.f,0.f,0.f,0.f,0.f,0.f,0.f};
        #pragma unroll
        for (int j = 0; j < 8; ++j) {
            const unsigned short* r0 = &ldsK[buf][lr][j * 32];
            const unsigned short* r1 = &ldsK[buf][16 + lr][j * 32];
            v16bf b0 = load_frag16(r0 + half * 8, r0 + 16 + half * 8);
            v16bf b1 = load_frag16(r1 + half * 8, r1 + 16 + half * 8);
            S0v = __builtin_amdgcn_wmma_f32_16x16x32_bf16(false, qf[j], false, b0,
                                                          (short)0, S0v, false, false);
            S1v = __builtin_amdgcn_wmma_f32_16x16x32_bf16(false, qf[j], false, b1,
                                                          (short)0, S1v, false, false);
        }

        // online softmax in exp2 domain; each row lives in one 16-lane group
        v8f corrv;
        #pragma unroll
        for (int r = 0; r < 8; ++r) {
            float s0f = S0v[r] * kScale;
            float s1f = S1v[r] * kScale;
            float rm = fmaxf(s0f, s1f);
            #pragma unroll
            for (int m = 1; m < 16; m <<= 1)
                rm = fmaxf(rm, __shfl_xor(rm, m, 32));
            float mnew = fmaxf(mrow[r], rm);
            float corr = fast_exp2(mrow[r] - mnew);
            float p0 = fast_exp2(s0f - mnew);
            float p1 = fast_exp2(s1f - mnew);
            float ps = p0 + p1;
            #pragma unroll
            for (int m = 1; m < 16; m <<= 1)
                ps += __shfl_xor(ps, m, 32);
            lrow[r] = lrow[r] * corr + ps;
            mrow[r] = mnew;
            corrv[r] = corr;
            unsigned pu = pk2(p0, p1);
            ldsP[wave][r + half * 8][lr]      = (unsigned short)pu;
            ldsP[wave][r + half * 8][16 + lr] = (unsigned short)(pu >> 16);
        }
        // vectorized rescale of all accumulators (packed f32 muls)
        #pragma unroll
        for (int t = 0; t < 16; ++t)
            acc[t] = acc[t] * corrv;

        // reread P as A fragment (same-wave LDS ops are in order)
        v16bf pf = load_frag16(&ldsP[wave][lr][half * 8],
                               &ldsP[wave][lr][16 + half * 8]);

        // acc[t] += P(16x32) * V(32x16) per 16-wide U tile; V from transposed LDS
        #pragma unroll
        for (int t = 0; t < 16; ++t) {
            const unsigned short* vr = &ldsVT[buf][t * 16 + lr][0];
            v16bf vf = load_frag16(vr + half * 8, vr + 16 + half * 8);
            acc[t] = __builtin_amdgcn_wmma_f32_16x16x32_bf16(false, pf, false, vf,
                                                             (short)0, acc[t], false, false);
        }
    }

    // normalize (vector reciprocal multiply) and store a = attn @ V (bf16)
    v8f linvv;
    #pragma unroll
    for (int r = 0; r < 8; ++r) linvv[r] = fast_rcp(lrow[r]);
    #pragma unroll
    for (int t = 0; t < 16; ++t) {
        v8f valv = acc[t] * linvv;
        #pragma unroll
        for (int r = 0; r < 8; r += 2) {
            int row = qRow0 + half * 8 + r;
            unsigned u = pk2(valv[r], valv[r + 1]);
            aout[((size_t)b * T_ + row) * U_ + t * 16 + lr]     = (unsigned short)u;
            aout[((size_t)b * T_ + row + 1) * U_ + t * 16 + lr] = (unsigned short)(u >> 16);
        }
    }
}

// ---------------------------------------------------------------------------
// Kernel 3: out = x + a @ Wa + ba.   [BT,U] x [U,C] -> fp32 [BT,C]
// ---------------------------------------------------------------------------
__global__ __launch_bounds__(256)
void out_proj_kernel(const unsigned short* __restrict__ a,
                     const float* __restrict__ Wa, const float* __restrict__ ba,
                     const float* __restrict__ x, float* __restrict__ out)
{
    __shared__ unsigned short ldsA[64][40];
    __shared__ unsigned short ldsWT[32][40];  // Wa transposed: [col][k]
    const int tid = threadIdx.x;
    const int lane = tid & 31, wave = tid >> 5;
    const int half = lane >> 4, lr = lane & 15;
    const int rowBase = blockIdx.x * 64;
    const int colBase = blockIdx.y * 32;
    const int wRow = (wave & 3) * 16;
    const int wCol = (wave >> 2) * 16;

    v8f acc = {0.f, 0.f, 0.f, 0.f, 0.f, 0.f, 0.f, 0.f};

    for (int k0 = 0; k0 < U_; k0 += 32) {
        // a tile is already bf16: async b128 copy (1 chunk per thread)
        {
            int r = tid >> 2, c8 = (tid & 3) * 8;
            async_copy_b128(lds_offset(&ldsA[r][c8]),
                            &a[(size_t)(rowBase + r) * U_ + k0 + c8]);
        }
        // Wa tile transposed, paired converts
        #pragma unroll
        for (int ii = 0; ii < 2; ++ii) {
            int idx = tid + ii * 256;
            int r = idx >> 4, c2 = (idx & 15) * 2;
            f32x2 wv = *(const f32x2*)&Wa[(size_t)(k0 + r) * C_ + colBase + c2];
            unsigned u = pk2(wv[0], wv[1]);
            ldsWT[c2][r]     = (unsigned short)u;
            ldsWT[c2 + 1][r] = (unsigned short)(u >> 16);
        }
        wait_asynccnt0();
        __syncthreads();

        v16bf af = load_frag16(&ldsA[wRow + lr][half * 8], &ldsA[wRow + lr][16 + half * 8]);
        v16bf bf = load_frag16(&ldsWT[wCol + lr][half * 8], &ldsWT[wCol + lr][16 + half * 8]);
        acc = __builtin_amdgcn_wmma_f32_16x16x32_bf16(false, af, false, bf,
                                                      (short)0, acc, false, false);
        __syncthreads();
    }

    const int col = colBase + wCol + lr;
    const float bval = ba[col];
    #pragma unroll
    for (int r = 0; r < 8; ++r) {
        int row = rowBase + wRow + half * 8 + r;
        out[(size_t)row * C_ + col] = x[(size_t)row * C_ + col] + acc[r] + bval;
    }
}

// ---------------------------------------------------------------------------
extern "C" void kernel_launch(void* const* d_in, const int* in_sizes, int n_in,
                              void* d_out, int out_size, void* d_ws, size_t ws_size,
                              hipStream_t stream)
{
    const float* x  = (const float*)d_in[0];
    const float* Wq = (const float*)d_in[1];
    const float* bq = (const float*)d_in[2];
    const float* Wk = (const float*)d_in[3];
    const float* bk = (const float*)d_in[4];
    const float* Wv = (const float*)d_in[5];
    const float* bv = (const float*)d_in[6];
    const float* Wa = (const float*)d_in[7];
    const float* ba = (const float*)d_in[8];
    float* out = (float*)d_out;

    const size_t n = (size_t)B_ * T_ * U_;
    unsigned short* qb = (unsigned short*)d_ws;
    unsigned short* kb = qb + n;
    unsigned short* vb = kb + n;
    unsigned short* ab = vb + n;

    dim3 blk(256);
    dim3 gP(B_ * T_ / 64, U_ / 32);
    qkv_proj_kernel<<<gP, blk, 0, stream>>>(x, Wq, bq, qb);
    qkv_proj_kernel<<<gP, blk, 0, stream>>>(x, Wk, bk, kb);
    qkv_proj_kernel<<<gP, blk, 0, stream>>>(x, Wv, bv, vb);

    dim3 gA(T_ / 128, B_);
    attn_kernel<<<gA, blk, 0, stream>>>(qb, kb, vb, ab);

    dim3 gO(B_ * T_ / 64, C_ / 32);
    out_proj_kernel<<<gO, blk, 0, stream>>>(ab, Wa, ba, x, out);
}
